// NoveltyDetector_55087250538839
// MI455X (gfx1250) — matrix operand order repaired and checked by reading, not compiled, not run on hardware
//
#include <hip/hip_runtime.h>
#include <hip/hip_bf16.h>

// ---------------------------------------------------------------------------
// NoveltyDetector for MI455X (gfx1250, wave32, WMMA, async-LDS)
//
//   h       = relu(x@W1+b1)         f32 WMMA 16x16x4   (precision-critical)
//   encoded = h@W2+b2               f32 WMMA 16x16x4   (checked output)
//   d2      = x2 + m2 - 2*enc@memT  f16 WMMA 16x16x32  (137 GFLOP, tanh-saturated)
//   out     = tanh(sqrt(max(min_col d2, 0)))   -- min taken on d2 (monotonic)
//
// Distance kernel pipelines memory tiles through LDS with
// GLOBAL_LOAD_ASYNC_TO_LDS_B128 (ASYNCcnt) and double buffering.
// ---------------------------------------------------------------------------

typedef __attribute__((ext_vector_type(16))) _Float16 v16h;
typedef __attribute__((ext_vector_type(4)))  _Float16 v4h;
typedef __attribute__((ext_vector_type(8)))  float    v8f;
typedef __attribute__((ext_vector_type(2)))  float    v2f;

#define B_    4096
#define IN_   256
#define HID_  1024
#define M_    65536

#define DBM   128     // distance kernel: batch rows per block
#define DBN   128     // distance kernel: memory rows per chunk
#define LSTR  264     // LDS row stride in halves: 528B = 16B-aligned, +4 banks/row

union Frag16 { v16h h; uint4 q[2]; };

// ---------------- Encoder GEMM1: h = relu(x @ W1 + b1), f32 WMMA -----------
__global__ __launch_bounds__(256) void enc1_kernel(const float* __restrict__ x,
                                                   const float* __restrict__ W1,
                                                   const float* __restrict__ b1,
                                                   float* __restrict__ h)
{
  const int lane  = threadIdx.x & 31;
  const int wid   = threadIdx.x >> 5;
  const int waveM = wid >> 2, waveN = wid & 3;         // 2 x 4 wave grid
  const int rowbase = blockIdx.x * 32 + waveM * 16;
  const int colbase = blockIdx.y * 64 + waveN * 16;
  const int row = rowbase + (lane & 15);
  const int col = colbase + (lane & 15);
  const int khi = (lane >> 4) * 2;                     // A/B lane K split for 16x16x4

  v8f acc = {};
  for (int kc = 0; kc < IN_ / 4; ++kc) {
    const int k0 = kc * 4 + khi;
    v2f a = *(const v2f*)(x + (size_t)row * IN_ + k0); // A: lane row, K pair
    v2f b;                                             // B: lane col, K pair
    b.x = W1[(size_t)k0 * HID_ + col];
    b.y = W1[(size_t)(k0 + 1) * HID_ + col];
    acc = __builtin_amdgcn_wmma_f32_16x16x4_f32(false, a, false, b, (short)0,
                                                acc, false, false);
  }
  const float bias = b1[col];
  const int rofs = (lane >> 4) * 8;
#pragma unroll
  for (int r = 0; r < 8; ++r) {
    float v = acc[r] + bias;
    v = v > 0.f ? v : 0.f;
    h[(size_t)(rowbase + rofs + r) * HID_ + col] = v;
  }
}

// ---------------- Encoder GEMM2: encoded = h @ W2 + b2, f32 WMMA -----------
__global__ __launch_bounds__(256) void enc2_kernel(const float* __restrict__ h,
                                                   const float* __restrict__ W2,
                                                   const float* __restrict__ b2,
                                                   float* __restrict__ enc_f32,
                                                   _Float16* __restrict__ enc_f16)
{
  const int lane  = threadIdx.x & 31;
  const int wid   = threadIdx.x >> 5;
  const int waveM = wid >> 2, waveN = wid & 3;
  const int rowbase = blockIdx.x * 32 + waveM * 16;
  const int colbase = blockIdx.y * 64 + waveN * 16;
  const int row = rowbase + (lane & 15);
  const int col = colbase + (lane & 15);
  const int khi = (lane >> 4) * 2;

  v8f acc = {};
  for (int kc = 0; kc < HID_ / 4; ++kc) {
    const int k0 = kc * 4 + khi;
    v2f a = *(const v2f*)(h + (size_t)row * HID_ + k0);
    v2f b;
    b.x = W2[(size_t)k0 * IN_ + col];
    b.y = W2[(size_t)(k0 + 1) * IN_ + col];
    acc = __builtin_amdgcn_wmma_f32_16x16x4_f32(false, a, false, b, (short)0,
                                                acc, false, false);
  }
  const float bias = b2[col];
  const int rofs = (lane >> 4) * 8;
#pragma unroll
  for (int r = 0; r < 8; ++r) {
    const float v = acc[r] + bias;
    const size_t idx = (size_t)(rowbase + rofs + r) * IN_ + col;
    enc_f32[idx] = v;
    enc_f16[idx] = (_Float16)v;
  }
}

// ---------------- f32 -> f16 bulk convert (memory matrix, once) ------------
__global__ __launch_bounds__(256) void cvt_kernel(const float* __restrict__ in,
                                                  _Float16* __restrict__ out)
{
  const size_t i = ((size_t)blockIdx.x * 256 + threadIdx.x) * 8;
  const float4 v0 = *(const float4*)(in + i);
  const float4 v1 = *(const float4*)(in + i + 4);
  v4h h0, h1;
  h0.x = (_Float16)v0.x; h0.y = (_Float16)v0.y; h0.z = (_Float16)v0.z; h0.w = (_Float16)v0.w;
  h1.x = (_Float16)v1.x; h1.y = (_Float16)v1.y; h1.z = (_Float16)v1.z; h1.w = (_Float16)v1.w;
  *(v4h*)(out + i)     = h0;
  *(v4h*)(out + i + 4) = h1;
}

// ---------------- row squared norms (f32) ----------------------------------
__global__ __launch_bounds__(256) void rownorm_kernel(const float* __restrict__ in,
                                                      float* __restrict__ out,
                                                      int cols)
{
  const int wid = threadIdx.x >> 5, lane = threadIdx.x & 31;
  const int row = blockIdx.x * 8 + wid;
  const float* p = in + (size_t)row * cols;
  float s = 0.f;
  for (int c = lane * 8; c < cols; c += 32 * 8) {
    const float4 a = *(const float4*)(p + c);
    const float4 b = *(const float4*)(p + c + 4);
    s += a.x*a.x + a.y*a.y + a.z*a.z + a.w*a.w;
    s += b.x*b.x + b.y*b.y + b.z*b.z + b.w*b.w;
  }
#pragma unroll
  for (int off = 16; off; off >>= 1) s += __shfl_xor(s, off, 32);
  if (lane == 0) out[row] = s;
}

// ---------------- init per-row min (+inf bits) -----------------------------
__global__ __launch_bounds__(256) void init_kernel(unsigned* __restrict__ minbits)
{
  minbits[blockIdx.x * 256 + threadIdx.x] = 0x7F800000u;
}

// ---------------- async global -> LDS helper (GV mode, B128) ---------------
__device__ __forceinline__ void async_g2l_b128(unsigned lds_byte_off, const void* gaddr)
{
  asm volatile("global_load_async_to_lds_b128 %0, %1, off"
               :: "v"(lds_byte_off), "v"((unsigned long long)(uintptr_t)gaddr)
               : "memory");
}
__device__ __forceinline__ void wait_async_zero()
{
  asm volatile("s_wait_asynccnt 0x0" ::: "memory");
}

// ---------------- fused distance GEMM + row-min, f16 WMMA ------------------
__global__ __launch_bounds__(256) void dist_kernel(const _Float16* __restrict__ enc,
                                                   const _Float16* __restrict__ memh,
                                                   const float* __restrict__ m2,
                                                   const float* __restrict__ x2,
                                                   unsigned* __restrict__ minbits)
{
  // LDS: A tile (DBM x LSTR) + double-buffered B tile (2 x DBN x LSTR), f16
  extern __shared__ _Float16 smem[];
  _Float16* Ald = smem;

  const int tid   = threadIdx.x;
  const int lane  = tid & 31, wid = tid >> 5;
  const int waveM = wid >> 1;            // 0..3 -> 32 rows each
  const int waveN = wid & 1;             // 0..1 -> 64 cols each
  const int l15   = lane & 15;
  const int hi    = lane >> 4;
  const int rowblock = blockIdx.x * DBM;

  const int nchunks = (M_ / DBN) / gridDim.y;
  const int c0 = blockIdx.y * nchunks;

  // ---- async-stage encoded tile (once) + first B tile ----
  for (int i = tid; i < DBM * (IN_ / 8); i += 256) {        // 16B granules
    const int row = i >> 5, kg = i & 31;
    async_g2l_b128((unsigned)((row * LSTR + kg * 8) * 2),
                   enc + (size_t)(rowblock + row) * IN_ + kg * 8);
  }
  {
    const int mbase = c0 * DBN;
    const unsigned bbase = (unsigned)(DBM * LSTR * 2);      // buffer 0
    for (int i = tid; i < DBN * (IN_ / 8); i += 256) {
      const int row = i >> 5, kg = i & 31;
      async_g2l_b128(bbase + (unsigned)((row * LSTR + kg * 8) * 2),
                     memh + (size_t)(mbase + row) * IN_ + kg * 8);
    }
  }
  wait_async_zero();
  __syncthreads();

  float minv[2][8];
#pragma unroll
  for (int t = 0; t < 2; ++t)
#pragma unroll
    for (int r = 0; r < 8; ++r) minv[t][r] = 3.4e38f;

  const v8f vzero = {};

  for (int ch = c0; ch < c0 + nchunks; ++ch) {
    const int buf = (ch - c0) & 1;
    const int mbase = ch * DBN;

    // prefetch next chunk into the other buffer (async, overlapped with WMMA)
    if (ch + 1 < c0 + nchunks) {
      const int nbase = (ch + 1) * DBN;
      const unsigned bbase = (unsigned)((DBM + (buf ^ 1) * DBN) * LSTR * 2);
      for (int i = tid; i < DBN * (IN_ / 8); i += 256) {
        const int row = i >> 5, kg = i & 31;
        async_g2l_b128(bbase + (unsigned)((row * LSTR + kg * 8) * 2),
                       memh + (size_t)(nbase + row) * IN_ + kg * 8);
      }
    }

    const _Float16* Bld = smem + (DBM + buf * DBN) * LSTR;

    float m2v[4];
#pragma unroll
    for (int u = 0; u < 4; ++u) m2v[u] = m2[mbase + waveN * 64 + u * 16 + l15];

    v8f acc[2][4];
#pragma unroll
    for (int t = 0; t < 2; ++t)
#pragma unroll
      for (int u = 0; u < 4; ++u) acc[t][u] = vzero;

#pragma unroll
    for (int kc = 0; kc < IN_ / 32; ++kc) {
      Frag16 a[2], b[4];
      // A fragments (ISA 16-bit A 16x32 layout): two b128 LDS reads each
#pragma unroll
      for (int t = 0; t < 2; ++t) {
        const _Float16* base = Ald + (waveM * 32 + t * 16 + l15) * LSTR
                                   + kc * 32 + hi * 8;
        a[t].q[0] = *(const uint4*)(base);
        a[t].q[1] = *(const uint4*)(base + 16);
      }
      // B fragments (lane = col; same K layout as A): two b128 reads each
#pragma unroll
      for (int u = 0; u < 4; ++u) {
        const _Float16* base = Bld + (waveN * 64 + u * 16 + l15) * LSTR
                                   + kc * 32 + hi * 8;
        b[u].q[0] = *(const uint4*)(base);
        b[u].q[1] = *(const uint4*)(base + 16);
      }
#pragma unroll
      for (int t = 0; t < 2; ++t)
#pragma unroll
        for (int u = 0; u < 4; ++u)
          acc[t][u] = __builtin_amdgcn_wmma_f32_16x16x32_f16(
              false, a[t].h, false, b[u].h, (short)0, acc[t][u], false, false);
    }

    // epilogue: min over columns of (m2 - 2*dot); x2 added at the very end
#pragma unroll
    for (int t = 0; t < 2; ++t)
#pragma unroll
      for (int u = 0; u < 4; ++u)
#pragma unroll
        for (int r = 0; r < 8; ++r)
          minv[t][r] = fminf(minv[t][r], m2v[u] - 2.0f * acc[t][u][r]);

    // next buffer's async loads must have landed; all waves done reading buf
    wait_async_zero();
    __syncthreads();
  }

  // reduce across the 16 lanes sharing each C row, then global atomic min
#pragma unroll
  for (int t = 0; t < 2; ++t) {
#pragma unroll
    for (int r = 0; r < 8; ++r) {
      float v = minv[t][r];
      v = fminf(v, __shfl_xor(v, 1, 32));
      v = fminf(v, __shfl_xor(v, 2, 32));
      v = fminf(v, __shfl_xor(v, 4, 32));
      v = fminf(v, __shfl_xor(v, 8, 32));
      if (l15 == 0) {
        const int row = rowblock + waveM * 32 + t * 16 + r + hi * 8;
        const float d2 = fmaxf(x2[row] + v, 0.0f);   // >=0 -> uint order == float order
        atomicMin(minbits + row, __float_as_uint(d2));
      }
    }
  }
}

// ---------------- finalize: tanh(sqrt(min d2)) -----------------------------
__global__ __launch_bounds__(256) void finish_kernel(const unsigned* __restrict__ minbits,
                                                     float* __restrict__ out)
{
  const int i = blockIdx.x * 256 + threadIdx.x;
  out[i] = tanhf(sqrtf(__uint_as_float(minbits[i])));
}

// ---------------------------------------------------------------------------
extern "C" void kernel_launch(void* const* d_in, const int* in_sizes, int n_in,
                              void* d_out, int out_size, void* d_ws, size_t ws_size,
                              hipStream_t stream)
{
  (void)in_sizes; (void)n_in; (void)out_size; (void)ws_size;
  const float* x   = (const float*)d_in[0];
  const float* W1  = (const float*)d_in[1];
  const float* b1  = (const float*)d_in[2];
  const float* W2  = (const float*)d_in[3];
  const float* b2  = (const float*)d_in[4];
  const float* mem = (const float*)d_in[5];

  char* ws = (char*)d_ws;
  float*    ws_h    = (float*)ws;                               // 16 MB
  _Float16* ws_enc  = (_Float16*)(ws + (16u << 20));            //  2 MB
  _Float16* ws_memh = (_Float16*)(ws + (18u << 20));            // 32 MB
  float*    ws_m2   = (float*)(ws + (50u << 20));               // 256 KB
  float*    ws_x2   = (float*)(ws + (50u << 20) + 262144);      //  16 KB
  unsigned* ws_min  = (unsigned*)(ws + (50u << 20) + 262144 + 16384);

  float* out     = (float*)d_out;     // [0,4096): novelty
  float* enc_f32 = out + B_;          // [4096, 4096+4096*256): encoded

  enc1_kernel<<<dim3(B_ / 32, HID_ / 64), 256, 0, stream>>>(x, W1, b1, ws_h);
  enc2_kernel<<<dim3(B_ / 32, IN_ / 64), 256, 0, stream>>>(ws_h, W2, b2, enc_f32, ws_enc);
  cvt_kernel<<<(M_ * IN_) / (256 * 8), 256, 0, stream>>>(mem, ws_memh);
  rownorm_kernel<<<B_ / 8, 256, 0, stream>>>(enc_f32, ws_x2, IN_);
  rownorm_kernel<<<M_ / 8, 256, 0, stream>>>(mem, ws_m2, IN_);
  init_kernel<<<B_ / 256, 256, 0, stream>>>(ws_min);

  const size_t smem = (size_t)(DBM + 2 * DBN) * LSTR * sizeof(_Float16); // 202,752 B
  dist_kernel<<<dim3(B_ / DBM, 16), 256, smem, stream>>>(ws_enc, ws_memh,
                                                         ws_m2, ws_x2, ws_min);
  finish_kernel<<<B_ / 256, 256, 0, stream>>>(ws_min, out);
}
